// Block_16698832847400
// MI455X (gfx1250) — matrix operand order repaired and checked by reading, not compiled, not run on hardware
//
#include <hip/hip_runtime.h>
#include <hip/hip_bf16.h>

// ---------------------------------------------------------------------------
// Types for CDNA5 WMMA (wave32)
// ---------------------------------------------------------------------------
typedef __attribute__((ext_vector_type(16))) __bf16 v16bf;
typedef __attribute__((ext_vector_type(8)))  float  v8f;
typedef __attribute__((ext_vector_type(4)))  int    v4i;

union Frag32B { uint4 u[2]; v16bf v; };

__device__ __forceinline__ unsigned short f2bf(float f) {
    unsigned u = __float_as_uint(f);
    unsigned r = (u + 0x7FFFu + ((u >> 16) & 1u)) >> 16;
    return (unsigned short)r;
}

// gfx1250 async global->LDS path (direct-to-LDS, tracked with ASYNCcnt)
// Probe confirmed: param0 is 'int4 addrspace(1)*' (prints as "int4 __device__ *")
#if defined(__gfx1250__) && __has_builtin(__builtin_amdgcn_global_load_async_to_lds_b128) && __has_builtin(__builtin_amdgcn_s_wait_asynccnt)
#define USE_ASYNC_LDS 1
typedef __attribute__((address_space(1))) v4i as1_v4i;
typedef __attribute__((address_space(3))) v4i as3_v4i;
#else
#define USE_ASYNC_LDS 0
#endif

// ---------------------------------------------------------------------------
// Constants for this block
// ---------------------------------------------------------------------------
#define BB   32
#define PP   14
#define WSZ  7
#define SSH  3
#define NWIN 49
#define NW4  4
#define NH   12
#define HD   64
#define CC   768
#define HID  3072
#define NTOK 196
#define MROWS 6272   // 32*196 == 128*49

// ---------------------------------------------------------------------------
// Generic bf16 WMMA GEMM:  C[M][N] = A[M][K] * W[N][K]^T (+ bias[n])
// M % 128 == 0, N % 128 == 0, K % 32 == 0.
// Block: 256 threads (8 waves, 4x2). WG tile 128x128, wave tile 32x64.
// Per K-step per wave: 8 wmma, 2 A frags + 4 B frags.
// ---------------------------------------------------------------------------
#define LDS_S 40   // padded LDS row stride in shorts (32 + 8)

__global__ __launch_bounds__(256)
void gemm_bf16(const unsigned short* __restrict__ A,
               const unsigned short* __restrict__ Bw,
               float* __restrict__ C,
               const float* __restrict__ bias,
               int M, int N, int K)
{
    __shared__ __align__(16) unsigned short As[128 * LDS_S];
    __shared__ __align__(16) unsigned short Bs[128 * LDS_S];

    const int t    = threadIdx.x;
    const int wave = t >> 5;
    const int lane = t & 31;
    const int wm   = wave & 3;       // 0..3  (M groups of 32)
    const int wn   = wave >> 2;      // 0..1  (N groups of 64)
    const int mBase = blockIdx.y * 128;
    const int nBase = blockIdx.x * 128;
    const int lrow  = lane & 15;
    const int hi    = lane >> 4;     // 0 or 1
    const int kbA   = hi ? 8 : 0;    // A frag first K run start
    const int ksB   = hi ? 8 : 0;    // B frag K run start

    // staging map: element l in [0,512): row = l>>2, col = (l&3)*8 (16B chunks)
    const int sRow = t >> 2;
    const int sCol = (t & 3) << 3;

    v8f acc[2][4];
    #pragma unroll
    for (int i = 0; i < 2; ++i)
        #pragma unroll
        for (int j = 0; j < 4; ++j) acc[i][j] = (v8f){0.f,0.f,0.f,0.f,0.f,0.f,0.f,0.f};

    for (int k0 = 0; k0 < K; k0 += 32) {
        // Stage A tile (128x32) and B tile (128 n-rows x 32 k): 2 x 16B each per thread
#if USE_ASYNC_LDS
        #pragma unroll
        for (int it = 0; it < 2; ++it) {
            int row = sRow + it * 64;
            __builtin_amdgcn_global_load_async_to_lds_b128(
                (as1_v4i*)(void*)(A + (size_t)(mBase + row) * K + k0 + sCol),
                (as3_v4i*)(void*)&As[row * LDS_S + sCol], 0, 0);
            __builtin_amdgcn_global_load_async_to_lds_b128(
                (as1_v4i*)(void*)(Bw + (size_t)(nBase + row) * K + k0 + sCol),
                (as3_v4i*)(void*)&Bs[row * LDS_S + sCol], 0, 0);
        }
        __builtin_amdgcn_s_wait_asynccnt(0);
#else
        #pragma unroll
        for (int it = 0; it < 2; ++it) {
            int row = sRow + it * 64;
            *(uint4*)&As[row * LDS_S + sCol] =
                *(const uint4*)(A + (size_t)(mBase + row) * K + k0 + sCol);
            *(uint4*)&Bs[row * LDS_S + sCol] =
                *(const uint4*)(Bw + (size_t)(nBase + row) * K + k0 + sCol);
        }
#endif
        __syncthreads();

        Frag32B af[2], bf_[4];
        #pragma unroll
        for (int mt = 0; mt < 2; ++mt) {
            int r = wm * 32 + mt * 16 + lrow;
            af[mt].u[0] = *(const uint4*)&As[r * LDS_S + kbA];
            af[mt].u[1] = *(const uint4*)&As[r * LDS_S + kbA + 16];
        }
        #pragma unroll
        for (int nt = 0; nt < 4; ++nt) {
            int n = wn * 64 + nt * 16 + lrow;
            bf_[nt].u[0] = *(const uint4*)&Bs[n * LDS_S + ksB];
            bf_[nt].u[1] = *(const uint4*)&Bs[n * LDS_S + ksB + 16];
        }

        #pragma unroll
        for (int mt = 0; mt < 2; ++mt)
            #pragma unroll
            for (int nt = 0; nt < 4; ++nt)
                acc[mt][nt] = __builtin_amdgcn_wmma_f32_16x16x32_bf16(
                    false, af[mt].v, false, bf_[nt].v,
                    (short)0, acc[mt][nt], false, false);
        __syncthreads();
    }

    // Epilogue: D layout -> VGPR r: lanes0-15 => (M=r, N=lane); lanes16-31 => (M=8+r, N=lane-16)
    #pragma unroll
    for (int mt = 0; mt < 2; ++mt) {
        #pragma unroll
        for (int nt = 0; nt < 4; ++nt) {
            int n = nBase + wn * 64 + nt * 16 + lrow;
            float bv = bias ? bias[n] : 0.0f;
            #pragma unroll
            for (int r = 0; r < 8; ++r) {
                int m = mBase + wm * 32 + mt * 16 + (hi ? 8 + r : r);
                C[(size_t)m * N + n] = acc[mt][nt][r] + bv;
            }
        }
    }
}

// ---------------------------------------------------------------------------
// fp32 -> bf16 conversion
// ---------------------------------------------------------------------------
__global__ void cvt_bf16(const float* __restrict__ in, unsigned short* __restrict__ out, size_t n) {
    size_t i = (size_t)blockIdx.x * blockDim.x + threadIdx.x;
    if (i < n) out[i] = f2bf(in[i]);
}

// ---------------------------------------------------------------------------
// Window gather: x[B,1+196,C] --roll(-3,-3)--> windows -> win_bf[6272][768]
// ---------------------------------------------------------------------------
__global__ void gather_win(const float* __restrict__ x, unsigned short* __restrict__ win) {
    int idx = blockIdx.x * blockDim.x + threadIdx.x;
    if (idx >= MROWS * CC) return;
    int c   = idx % CC;
    int row = idx / CC;
    int w   = row / NWIN;
    int i   = row % NWIN;
    int b   = w / NW4;
    int nw  = w % NW4;
    int wy = nw >> 1, wx = nw & 1;
    int ly = i / WSZ, lx = i % WSZ;
    int sy = wy * WSZ + ly, sx = wx * WSZ + lx;
    int y = (sy + SSH) % PP, xx = (sx + SSH) % PP;
    int tok = y * PP + xx;
    win[idx] = f2bf(x[((size_t)(b * 197 + 1 + tok)) * CC + c]);
}

// ---------------------------------------------------------------------------
// Windowed attention with inline rel-pos bias + shift mask + softmax.
// grid = 128 windows * 12 heads; block = 256.
// ---------------------------------------------------------------------------
__device__ __forceinline__ int rcat(int v) { return v < WSZ ? 0 : (v < (PP - SSH) ? 1 : 2); }

__global__ __launch_bounds__(256)
void attn_kernel(const float* __restrict__ qkv, const float* __restrict__ rpb,
                 unsigned short* __restrict__ ao)
{
    __shared__ float qs[NWIN * HD];
    __shared__ float ks_[NWIN * HD];
    __shared__ float vs[NWIN * HD];
    __shared__ float sc[NWIN * NWIN];

    const int w  = blockIdx.x / NH;
    const int h  = blockIdx.x % NH;
    const int nw = w & 3;
    const int wy = nw >> 1, wx = nw & 1;
    const int t  = threadIdx.x;

    for (int e = t; e < NWIN * HD; e += 256) {
        int i = e >> 6, d = e & 63;
        size_t base = (size_t)(w * NWIN + i) * (3 * CC) + h * HD + d;
        qs[e]  = qkv[base];
        ks_[e] = qkv[base + CC];
        vs[e]  = qkv[base + 2 * CC];
    }
    __syncthreads();

    for (int e = t; e < NWIN * NWIN; e += 256) {
        int i = e / NWIN, j = e % NWIN;
        float dot = 0.f;
        #pragma unroll 8
        for (int d = 0; d < HD; ++d) dot += qs[i * HD + d] * ks_[j * HD + d];
        int yi = i / WSZ, xi = i % WSZ, yj = j / WSZ, xj = j % WSZ;
        int rpi = (yi - yj + WSZ - 1) * (2 * WSZ - 1) + (xi - xj + WSZ - 1);
        float bias = rpb[rpi * NH + h];
        int ri = rcat(wy * WSZ + yi) * 3 + rcat(wx * WSZ + xi);
        int rj = rcat(wy * WSZ + yj) * 3 + rcat(wx * WSZ + xj);
        float mask = (ri == rj) ? 0.f : -100.f;
        sc[e] = 0.125f * dot + bias + mask;
    }
    __syncthreads();

    if (t < NWIN) {
        float* row = &sc[t * NWIN];
        float mx = row[0];
        for (int j = 1; j < NWIN; ++j) mx = fmaxf(mx, row[j]);
        float s = 0.f;
        for (int j = 0; j < NWIN; ++j) { float e = __expf(row[j] - mx); row[j] = e; s += e; }
        float inv = 1.f / s;
        for (int j = 0; j < NWIN; ++j) row[j] *= inv;
    }
    __syncthreads();

    for (int e = t; e < NWIN * HD; e += 256) {
        int i = e >> 6, d = e & 63;
        float s = 0.f;
        #pragma unroll 7
        for (int j = 0; j < NWIN; ++j) s += sc[i * NWIN + j] * vs[j * HD + d];
        ao[(size_t)(w * NWIN + i) * CC + h * HD + d] = f2bf(s);
    }
}

// ---------------------------------------------------------------------------
// Un-window + roll(+3,+3): proj_out[6272][768] -> f[B*196][768] (f32 + bf16)
// ---------------------------------------------------------------------------
__global__ void unshift(const float* __restrict__ proj, float* __restrict__ f32,
                        unsigned short* __restrict__ fbf) {
    int idx = blockIdx.x * blockDim.x + threadIdx.x;
    if (idx >= MROWS * CC) return;
    int c   = idx % CC;
    int row = idx / CC;
    int b = row / NTOK, tk = row % NTOK;
    int y = tk / PP, x = tk % PP;
    int sy = (y + PP - SSH) % PP, sx = (x + PP - SSH) % PP;
    int wy = sy / WSZ, wx = sx / WSZ;
    int nw = wy * 2 + wx;
    int i  = (sy % WSZ) * WSZ + (sx % WSZ);
    int srow = (b * NW4 + nw) * NWIN + i;
    float v = proj[(size_t)srow * CC + c];
    f32[idx] = v;
    fbf[idx] = f2bf(v);
}

// ---------------------------------------------------------------------------
// BN batch stats -> scale/shift.  grid.x = Cc/64, block = 256 (64c x 4 slices)
// ---------------------------------------------------------------------------
__global__ __launch_bounds__(256)
void bnstats(const float* __restrict__ X, const float* __restrict__ g,
             const float* __restrict__ b, float* __restrict__ scale,
             float* __restrict__ shift, int R, int Cc)
{
    __shared__ float ss[256], sq[256];
    int t = threadIdx.x;
    int c = blockIdx.x * 64 + (t & 63);
    int sl = t >> 6;
    float s = 0.f, q = 0.f;
    for (int r = sl; r < R; r += 4) {
        float v = X[(size_t)r * Cc + c];
        s += v; q += v * v;
    }
    ss[t] = s; sq[t] = q;
    __syncthreads();
    if (t < 64) {
        s = ss[t] + ss[t + 64] + ss[t + 128] + ss[t + 192];
        q = sq[t] + sq[t + 64] + sq[t + 128] + sq[t + 192];
        float mean = s / (float)R;
        float var  = q / (float)R - mean * mean;
        float sc   = g[c] * rsqrtf(var + 1e-5f);
        scale[c] = sc;
        shift[c] = b[c] - mean * sc;
    }
}

__global__ void bn_hswish(float* __restrict__ X, const float* __restrict__ scale,
                          const float* __restrict__ shift, size_t n, int Cc) {
    size_t i = (size_t)blockIdx.x * blockDim.x + threadIdx.x;
    if (i >= n) return;
    int c = (int)(i % Cc);
    float x = X[i] * scale[c] + shift[c];
    X[i] = x * fminf(fmaxf(x + 3.f, 0.f), 6.f) * (1.f / 6.f);
}

// ---------------------------------------------------------------------------
// 3x3 depthwise conv, SAME pad, NHWC-ish [B*196][3072]
// ---------------------------------------------------------------------------
__global__ void dwconv(const float* __restrict__ Hin, const float* __restrict__ Wd,
                       float* __restrict__ Hout) {
    int idx = blockIdx.x * blockDim.x + threadIdx.x;
    if (idx >= BB * NTOK * HID) return;
    int c = idx % HID;
    int tk = (idx / HID) % NTOK;
    int b  = idx / (HID * NTOK);
    int y = tk / PP, x = tk % PP;
    float s = 0.f;
    #pragma unroll
    for (int dy = 0; dy < 3; ++dy) {
        int yy = y + dy - 1;
        if (yy < 0 || yy >= PP) continue;
        #pragma unroll
        for (int dx = 0; dx < 3; ++dx) {
            int xx = x + dx - 1;
            if (xx < 0 || xx >= PP) continue;
            s += Hin[((size_t)(b * NTOK + yy * PP + xx)) * HID + c] * Wd[c * 9 + dy * 3 + dx];
        }
    }
    Hout[idx] = s;
}

// ---------------------------------------------------------------------------
// SE: spatial mean, tiny FCs, channel scale (+ bf16 convert for pw2)
// ---------------------------------------------------------------------------
__global__ __launch_bounds__(256)
void se_mean(const float* __restrict__ H, float* __restrict__ sm) {
    __shared__ float ss[256];
    int t = threadIdx.x;
    int c = blockIdx.x * 64 + (t & 63);
    int b = blockIdx.y;
    int sl = t >> 6;
    float s = 0.f;
    for (int r = sl; r < NTOK; r += 4) s += H[((size_t)(b * NTOK + r)) * HID + c];
    ss[t] = s;
    __syncthreads();
    if (t < 64)
        sm[b * HID + c] = (ss[t] + ss[t + 64] + ss[t + 128] + ss[t + 192]) * (1.f / (float)NTOK);
}

__global__ void fc_kernel(const float* __restrict__ in, const float* __restrict__ W,
                          const float* __restrict__ bias, float* __restrict__ out,
                          int Bn, int In, int On, int act) {
    int idx = blockIdx.x * blockDim.x + threadIdx.x;
    if (idx >= Bn * On) return;
    int b = idx / On, o = idx % On;
    float s = bias[o];
    const float* ip = in + (size_t)b * In;
    const float* wp = W + (size_t)o * In;
    for (int k = 0; k < In; ++k) s += ip[k] * wp[k];
    out[idx] = (act == 0) ? fmaxf(s, 0.f)
                          : fminf(fmaxf(s + 3.f, 0.f), 6.f) * (1.f / 6.f);
}

__global__ void se_scale_bf16(const float* __restrict__ H, const float* __restrict__ s2,
                              unsigned short* __restrict__ out) {
    int idx = blockIdx.x * blockDim.x + threadIdx.x;
    if (idx >= BB * NTOK * HID) return;
    int c = idx % HID;
    int b = idx / (HID * NTOK);
    out[idx] = f2bf(H[idx] * s2[b * HID + c]);
}

// ---------------------------------------------------------------------------
// Final: out = concat(cls, f + bn3(g))
// ---------------------------------------------------------------------------
__global__ void final_assemble(const float* __restrict__ xin, const float* __restrict__ f,
                               const float* __restrict__ g, const float* __restrict__ sc3,
                               const float* __restrict__ sh3, float* __restrict__ out) {
    int idx = blockIdx.x * blockDim.x + threadIdx.x;
    if (idx >= BB * 197 * CC) return;
    int c = idx % CC;
    int t197 = (idx / CC) % 197;
    int b = idx / (CC * 197);
    if (t197 == 0) { out[idx] = xin[idx]; return; }
    size_t r = (size_t)(b * NTOK + t197 - 1);
    out[idx] = f[r * CC + c] + (g[r * CC + c] * sc3[c] + sh3[c]);
}

// ---------------------------------------------------------------------------
// Launch
// ---------------------------------------------------------------------------
extern "C" void kernel_launch(void* const* d_in, const int* in_sizes, int n_in,
                              void* d_out, int out_size, void* d_ws, size_t ws_size,
                              hipStream_t stream) {
    const float* x      = (const float*)d_in[0];
    const float* w_qkv  = (const float*)d_in[1];
    const float* w_proj = (const float*)d_in[2];
    const float* b_proj = (const float*)d_in[3];
    const float* rpb    = (const float*)d_in[4];
    const float* pw1_w  = (const float*)d_in[5];
    const float* bn1_g  = (const float*)d_in[6];
    const float* bn1_b  = (const float*)d_in[7];
    const float* dw_w   = (const float*)d_in[8];
    const float* bn2_g  = (const float*)d_in[9];
    const float* bn2_b  = (const float*)d_in[10];
    const float* se_w1  = (const float*)d_in[11];
    const float* se_b1  = (const float*)d_in[12];
    const float* se_w2  = (const float*)d_in[13];
    const float* se_b2  = (const float*)d_in[14];
    const float* pw2_w  = (const float*)d_in[15];
    const float* bn3_g  = (const float*)d_in[16];
    const float* bn3_b  = (const float*)d_in[17];
    float* out = (float*)d_out;

    char* p = (char*)d_ws;
    auto alloc = [&](size_t bytes) -> char* {
        char* r = p;
        p += (bytes + 255) & ~(size_t)255;
        return r;
    };

    unsigned short* wq_bf  = (unsigned short*)alloc((size_t)3 * CC * CC * 2); // 2304*768
    unsigned short* wp_bf  = (unsigned short*)alloc((size_t)CC * CC * 2);
    unsigned short* w1_bf  = (unsigned short*)alloc((size_t)HID * CC * 2);
    unsigned short* w2_bf  = (unsigned short*)alloc((size_t)CC * HID * 2);
    unsigned short* win_bf = (unsigned short*)alloc((size_t)MROWS * CC * 2);
    float*          qkv_f  = (float*)alloc((size_t)MROWS * 3 * CC * 4);
    unsigned short* ao_bf  = (unsigned short*)alloc((size_t)MROWS * CC * 2);
    float*          proj_f = (float*)alloc((size_t)MROWS * CC * 4);
    float*          f_f32  = (float*)alloc((size_t)MROWS * CC * 4);
    unsigned short* f_bf   = (unsigned short*)alloc((size_t)MROWS * CC * 2);
    float*          h1     = (float*)alloc((size_t)MROWS * HID * 4);
    float*          h2     = (float*)alloc((size_t)MROWS * HID * 4);

    // aliases into dead regions:
    unsigned short* h3s_bf = (unsigned short*)qkv_f;                    // after attention
    float* g_f   = (float*)((char*)qkv_f + (size_t)MROWS * HID * 2);    // 38.5MB in
    float* s_mean = (float*)win_bf;                                     // after QKV gemm
    float* s1  = (float*)((char*)win_bf + 393216);
    float* s2  = (float*)((char*)win_bf + 491520);
    float* sc1 = (float*)((char*)win_bf + 884736);
    float* sh1 = (float*)((char*)win_bf + 897024);
    float* sc2 = (float*)((char*)win_bf + 909312);
    float* sh2 = (float*)((char*)win_bf + 921600);
    float* sc3 = (float*)((char*)win_bf + 933888);
    float* sh3 = (float*)((char*)win_bf + 937216);

    auto gb = [](size_t n) { return (unsigned)((n + 255) / 256); };

    // weight conversions
    cvt_bf16<<<gb((size_t)3*CC*CC), 256, 0, stream>>>(w_qkv, wq_bf, (size_t)3*CC*CC);
    cvt_bf16<<<gb((size_t)CC*CC), 256, 0, stream>>>(w_proj, wp_bf, (size_t)CC*CC);
    cvt_bf16<<<gb((size_t)HID*CC), 256, 0, stream>>>(pw1_w, w1_bf, (size_t)HID*CC);
    cvt_bf16<<<gb((size_t)CC*HID), 256, 0, stream>>>(pw2_w, w2_bf, (size_t)CC*HID);

    // attention path
    gather_win<<<gb((size_t)MROWS*CC), 256, 0, stream>>>(x, win_bf);
    gemm_bf16<<<dim3((3*CC)/128, MROWS/128), 256, 0, stream>>>(win_bf, wq_bf, qkv_f, nullptr, MROWS, 3*CC, CC);
    attn_kernel<<<128 * NH, 256, 0, stream>>>(qkv_f, rpb, ao_bf);
    gemm_bf16<<<dim3(CC/128, MROWS/128), 256, 0, stream>>>(ao_bf, wp_bf, proj_f, b_proj, MROWS, CC, CC);
    unshift<<<gb((size_t)MROWS*CC), 256, 0, stream>>>(proj_f, f_f32, f_bf);

    // MLP path
    gemm_bf16<<<dim3(HID/128, MROWS/128), 256, 0, stream>>>(f_bf, w1_bf, h1, nullptr, MROWS, HID, CC);
    bnstats<<<HID/64, 256, 0, stream>>>(h1, bn1_g, bn1_b, sc1, sh1, MROWS, HID);
    bn_hswish<<<gb((size_t)MROWS*HID), 256, 0, stream>>>(h1, sc1, sh1, (size_t)MROWS*HID, HID);
    dwconv<<<gb((size_t)MROWS*HID), 256, 0, stream>>>(h1, dw_w, h2);
    bnstats<<<HID/64, 256, 0, stream>>>(h2, bn2_g, bn2_b, sc2, sh2, MROWS, HID);
    bn_hswish<<<gb((size_t)MROWS*HID), 256, 0, stream>>>(h2, sc2, sh2, (size_t)MROWS*HID, HID);

    // SE
    se_mean<<<dim3(HID/64, BB), 256, 0, stream>>>(h2, s_mean);
    fc_kernel<<<gb((size_t)BB*(HID/4)), 256, 0, stream>>>(s_mean, se_w1, se_b1, s1, BB, HID, HID/4, 0);
    fc_kernel<<<gb((size_t)BB*HID), 256, 0, stream>>>(s1, se_w2, se_b2, s2, BB, HID/4, HID, 1);
    se_scale_bf16<<<gb((size_t)MROWS*HID), 256, 0, stream>>>(h2, s2, h3s_bf);

    // pw2 + bn3 + residual + cls
    gemm_bf16<<<dim3(CC/128, MROWS/128), 256, 0, stream>>>(h3s_bf, w2_bf, g_f, nullptr, MROWS, CC, HID);
    bnstats<<<CC/64, 256, 0, stream>>>(g_f, bn3_g, bn3_b, sc3, sh3, MROWS, CC);
    final_assemble<<<gb((size_t)BB*197*CC), 256, 0, stream>>>(x, f_f32, g_f, sc3, sh3, out);
}
